// SLACLatentModel_18365280158081
// MI455X (gfx1250) — compile-verified, matrix operand order checked by reading, not run on hardware
//
#include <hip/hip_runtime.h>
#include <math.h>

// ---------------------------------------------------------------------------
// SLAC latent model scan on gfx1250 (wave32, WMMA).
// Each workgroup = 256 threads (8 waves) owns 16 batch rows and runs the full
// T=64 sequential scan with the (z1, z2) carry kept in LDS as f16.
// All matmuls are M=16 WMMA tiles: v_wmma_f32_16x16x32_f16, f32 accumulate.
// Weights are repacked once (device-side) into the wave32 B-fragment layout.
// ---------------------------------------------------------------------------

typedef __attribute__((ext_vector_type(16))) _Float16 v16h;
typedef __attribute__((ext_vector_type(8)))  float    v8f;

#define BATCH 1024
#define TT    64
#define EE    256
#define AA    6
#define ZZ1   32
#define ZZ2   256
#define OUTC  416   // z1(32)|z2(256)|qm(32)|qs(32)|pm(32)|ps(32)

// Repacked-weight offsets in d_ws (units: unsigned short / f16 elements).
// size = Kpad * N  (Kpad = ceil(K/32)*32)
constexpr unsigned OFF_I1W1 = 0;        // 256x256
constexpr unsigned OFF_I1W2 = 65536;    // 256x256
constexpr unsigned OFF_I1W3 = 131072;   // 256x64
constexpr unsigned OFF_I2W1 = 147456;   // 32x256
constexpr unsigned OFF_I2W2 = 155648;   // 256x256
constexpr unsigned OFF_I2W3 = 221184;   // 256x512
constexpr unsigned OFF_POW1 = 352256;   // 544x256 (K=518 padded)
constexpr unsigned OFF_POW2 = 491520;   // 256x256
constexpr unsigned OFF_POW3 = 557056;   // 256x64
constexpr unsigned OFF_Z2W1 = 573440;   // 320x256 (K=294 padded)
constexpr unsigned OFF_Z2W2 = 655360;   // 256x256
constexpr unsigned OFF_Z2W3 = 720896;   // 256x512
constexpr unsigned OFF_PRW1 = 851968;   // 288x256 (K=262 padded)
constexpr unsigned OFF_PRW2 = 925696;   // 256x256
constexpr unsigned OFF_PRW3 = 991232;   // 256x64
// total = 1,007,616 ushorts = ~2.02 MB of d_ws

struct Params {
    const float *emb, *act, *eps1, *eps2;
    const float *b_i1_1, *b_i1_2, *b_i1_3;
    const float *b_i2_1, *b_i2_2, *b_i2_3;
    const float *b_po_1, *b_po_2, *b_po_3;
    const float *b_z2_1, *b_z2_2, *b_z2_3;
    const float *b_pr_1, *b_pr_2, *b_pr_3;
    const unsigned short *ws;
    float *out;
};

// ---------------------------------------------------------------------------
// Weight repack: f32 W[K][N] row-major  ->  wave32 WMMA B-fragments (f16).
// Fragment = one 32(K)x16(N) slab: 32 lanes x 8 dwords (each dword = K,K+1
// f16 pair for one N column). Fragment order: ntile-major, then K-slab,
// matching the GEMM inner loop so each lane does 2x global_load_b128.
// ---------------------------------------------------------------------------
__global__ void repack_kernel(const float* __restrict__ W,
                              unsigned short* __restrict__ dst,
                              int K, int N, int Kpad) {
    int nslab = Kpad >> 5;
    int total = (Kpad >> 1) * N;                 // dwords
    int tid = blockIdx.x * 256 + threadIdx.x;
    if (tid >= total) return;
    int f    = tid >> 8;                         // 256 dwords / fragment
    int r    = tid & 255;
    int lane = r >> 3;
    int v    = r & 7;
    int ntile = f / nslab;
    int slab  = f - ntile * nslab;
    int n  = ntile * 16 + (lane & 15);
    int k0 = slab * 32 + ((lane & 16) ? 8 : 0) + ((v & 3) << 1) + ((v >> 2) << 4);
    _Float16 lo = (k0     < K) ? (_Float16)W[(size_t)k0 * N + n]     : (_Float16)0.f;
    _Float16 hi = (k0 + 1 < K) ? (_Float16)W[(size_t)(k0 + 1) * N + n] : (_Float16)0.f;
    union { _Float16 h[2]; unsigned int u; } pk;
    pk.h[0] = lo; pk.h[1] = hi;
    ((unsigned int*)dst)[f * 256 + lane * 8 + v] = pk.u;
}

static __device__ __forceinline__ float leakyf(float x) {
    return x > 0.f ? x : 0.2f * x;
}
static __device__ __forceinline__ float softplusf(float x) {
    return x > 20.f ? x : log1pf(expf(x));
}

// One 16x16 output tile accumulated over nslab K-slabs of 32.
// A from LDS f16 activations (row-major, leading dim ldx), B from repacked
// global fragments. Emits v_wmma_f32_16x16x32_f16.
static __device__ __forceinline__ v8f gemm_f16(const _Float16* xl, int ldx,
                                               const unsigned short* wrep,
                                               int nslab, int lane) {
    v8f acc = {};
    const int m  = lane & 15;
    const int kb = (lane & 16) ? 8 : 0;
    for (int s = 0; s < nslab; ++s) {
        union { v16h v; unsigned int u[8]; } A;
        const _Float16* xr = xl + m * ldx + s * 32;
#pragma unroll
        for (int v = 0; v < 8; ++v) {
            int koff = kb + ((v & 3) << 1) + ((v >> 2) << 4);
            A.u[v] = *(const unsigned int*)(xr + koff);   // ds_load_b32 (K,K+1)
        }
        union { v16h v; uint4 q[2]; } Bu;
        const uint4* wp = (const uint4*)(wrep + (size_t)s * 512 + lane * 16);
        Bu.q[0] = wp[0];                                   // global_load_b128
        Bu.q[1] = wp[1];
        acc = __builtin_amdgcn_wmma_f32_16x16x32_f16(false, A.v, false, Bu.v,
                                                     (short)0, acc, false, false);
    }
    return acc;
}

// Hidden layer: [16 x Kpad] @ [Kpad x 256] + bias, leaky(0.2), store f16 to LDS.
// 16 N-tiles split 2-per-wave across 8 waves.
static __device__ __forceinline__ void dense_layer(const _Float16* xl, int Kpad,
                                                   const unsigned short* wrep,
                                                   const float* bias,
                                                   _Float16* dst,
                                                   int wave, int lane) {
    const int nslab = Kpad >> 5;
    const int mb = (lane & 16) ? 8 : 0;
    for (int nt = wave * 2; nt < wave * 2 + 2; ++nt) {
        v8f acc = gemm_f16(xl, Kpad, wrep + (size_t)nt * nslab * 512, nslab, lane);
        int n = nt * 16 + (lane & 15);
        float bv = bias[n];
#pragma unroll
        for (int i = 0; i < 8; ++i) {
            float r = leakyf(acc[i] + bv);
            dst[(mb + i) * 256 + n] = (_Float16)r;         // ds_store_b16
        }
    }
}

// z1-style head (2*Z1 = 64 outputs): waves 0..1 each own one (mean,raw) column
// pair; fuses softplus + reparameterization + global stores + LDS carry.
static __device__ __forceinline__ void head_z1(const _Float16* h2,
                                               const unsigned short* w3,
                                               const float* b3,
                                               const float* eps1, float* out,
                                               _Float16* z1s, int b0, int t,
                                               int wave, int lane,
                                               bool posterior,
                                               int col_m, int col_s) {
    if (wave < 2) {
        const int p  = wave;
        const int mb = (lane & 16) ? 8 : 0;
        v8f am = gemm_f16(h2, 256, w3 + (size_t)p * 8 * 512, 8, lane);
        v8f ar = gemm_f16(h2, 256, w3 + (size_t)(p + 2) * 8 * 512, 8, lane);
        int n = p * 16 + (lane & 15);
        float bm = b3[n], br = b3[32 + n];
#pragma unroll
        for (int i = 0; i < 8; ++i) {
            int m = mb + i;
            int b = b0 + m;
            float qm = am[i] + bm;
            float qs = softplusf(ar[i] + br) + 1e-5f;
            size_t ob = ((size_t)b * TT + t) * OUTC;
            out[ob + col_m + n] = qm;
            out[ob + col_s + n] = qs;
            if (posterior) {
                float e = eps1[((size_t)b * TT + t) * ZZ1 + n];
                float z = qm + qs * e;
                out[ob + 0 + n] = z;
                z1s[m * ZZ1 + n] = (_Float16)z;
            }
        }
    }
}

// z2 head (2*Z2D = 512 outputs): 16 (mean,raw) column pairs, 2 per wave.
static __device__ __forceinline__ void head_z2(const _Float16* h2,
                                               const unsigned short* w3,
                                               const float* b3,
                                               const float* eps2, float* out,
                                               _Float16* z2s, int b0, int t,
                                               int wave, int lane) {
    const int mb = (lane & 16) ? 8 : 0;
    for (int p = wave * 2; p < wave * 2 + 2; ++p) {
        v8f am = gemm_f16(h2, 256, w3 + (size_t)p * 8 * 512, 8, lane);
        v8f ar = gemm_f16(h2, 256, w3 + (size_t)(p + 16) * 8 * 512, 8, lane);
        int n = p * 16 + (lane & 15);
        float bm = b3[n], br = b3[256 + n];
#pragma unroll
        for (int i = 0; i < 8; ++i) {
            int m = mb + i;
            int b = b0 + m;
            float mu = am[i] + bm;
            float sd = softplusf(ar[i] + br) + 1e-5f;
            float z = mu + sd * eps2[((size_t)b * TT + t) * ZZ2 + n];
            out[((size_t)b * TT + t) * OUTC + 32 + n] = z;
            z2s[m * ZZ2 + n] = (_Float16)z;
        }
    }
}

__global__ __launch_bounds__(256) void slac_main(Params p) {
    const int tid  = threadIdx.x;
    const int wave = tid >> 5;
    const int lane = tid & 31;
    const int b0   = blockIdx.x * 16;

    __shared__ _Float16 xbuf[16 * 544];   // layer input (max K-pad 544)
    __shared__ _Float16 hbuf[16 * 256];
    __shared__ _Float16 h2buf[16 * 256];
    __shared__ _Float16 z1s[16 * ZZ1];    // carry
    __shared__ _Float16 z2s[16 * ZZ2];    // carry

    // ---------------- t = 0: posterior_init + z2_init, fixed prior ----------
    for (int i = tid; i < 16 * 256; i += 256) {
        int m = i >> 8, c = i & 255;
        xbuf[i] = (_Float16)p.emb[((size_t)(b0 + m) * TT + 0) * EE + c];
    }
    __syncthreads();
    dense_layer(xbuf, 256, p.ws + OFF_I1W1, p.b_i1_1, hbuf, wave, lane);
    __syncthreads();
    dense_layer(hbuf, 256, p.ws + OFF_I1W2, p.b_i1_2, h2buf, wave, lane);
    __syncthreads();
    head_z1(h2buf, p.ws + OFF_I1W3, p.b_i1_3, p.eps1, p.out, z1s,
            b0, 0, wave, lane, true, 288, 320);
    for (int i = tid; i < 16 * ZZ1; i += 256) {          // FixedGaussian prior
        int m = i >> 5, n = i & 31;
        size_t ob = ((size_t)(b0 + m) * TT + 0) * OUTC;
        p.out[ob + 352 + n] = 0.f;
        p.out[ob + 384 + n] = 1.f;
    }
    __syncthreads();
    for (int i = tid; i < 16 * ZZ1; i += 256) xbuf[i] = z1s[i];
    __syncthreads();
    dense_layer(xbuf, 32, p.ws + OFF_I2W1, p.b_i2_1, hbuf, wave, lane);
    __syncthreads();
    dense_layer(hbuf, 256, p.ws + OFF_I2W2, p.b_i2_2, h2buf, wave, lane);
    __syncthreads();
    head_z2(h2buf, p.ws + OFF_I2W3, p.b_i2_3, p.eps2, p.out, z2s, b0, 0, wave, lane);
    __syncthreads();

    // ---------------- t = 1..63 scan ----------------------------------------
    for (int t = 1; t < TT; ++t) {
        // x_post = [z2_old(256) | a_prev(6) | emb_t(256)]  K=518 -> 544
        for (int i = tid; i < 16 * 544; i += 256) {
            int m = i / 544, c = i - m * 544;
            _Float16 h;
            if (c < 256)      h = z2s[m * ZZ2 + c];
            else if (c < 262) h = (_Float16)p.act[((size_t)(b0 + m) * (TT - 1) + (t - 1)) * AA + (c - 256)];
            else if (c < 518) h = (_Float16)p.emb[((size_t)(b0 + m) * TT + t) * EE + (c - 262)];
            else              h = (_Float16)0.f;
            xbuf[i] = h;
        }
        __syncthreads();
        dense_layer(xbuf, 544, p.ws + OFF_POW1, p.b_po_1, hbuf, wave, lane);
        __syncthreads();
        dense_layer(hbuf, 256, p.ws + OFF_POW2, p.b_po_2, h2buf, wave, lane);
        __syncthreads();
        head_z1(h2buf, p.ws + OFF_POW3, p.b_po_3, p.eps1, p.out, z1s,
                b0, t, wave, lane, true, 288, 320);
        __syncthreads();

        // x_z2 = [z1_new(32) | z2_old(256) | a_prev(6)]  K=294 -> 320
        for (int i = tid; i < 16 * 320; i += 256) {
            int m = i / 320, c = i - m * 320;
            _Float16 h;
            if (c < 32)       h = z1s[m * ZZ1 + c];
            else if (c < 288) h = z2s[m * ZZ2 + (c - 32)];
            else if (c < 294) h = (_Float16)p.act[((size_t)(b0 + m) * (TT - 1) + (t - 1)) * AA + (c - 288)];
            else              h = (_Float16)0.f;
            xbuf[i] = h;
        }
        __syncthreads();
        dense_layer(xbuf, 320, p.ws + OFF_Z2W1, p.b_z2_1, hbuf, wave, lane);
        __syncthreads();
        dense_layer(hbuf, 256, p.ws + OFF_Z2W2, p.b_z2_2, h2buf, wave, lane);
        __syncthreads();
        head_z2(h2buf, p.ws + OFF_Z2W3, p.b_z2_3, p.eps2, p.out, z2s, b0, t, wave, lane);
        __syncthreads();

        // x_prior = [z2_new(256) | a_prev(6)]  K=262 -> 288
        for (int i = tid; i < 16 * 288; i += 256) {
            int m = i / 288, c = i - m * 288;
            _Float16 h;
            if (c < 256)      h = z2s[m * ZZ2 + c];
            else if (c < 262) h = (_Float16)p.act[((size_t)(b0 + m) * (TT - 1) + (t - 1)) * AA + (c - 256)];
            else              h = (_Float16)0.f;
            xbuf[i] = h;
        }
        __syncthreads();
        dense_layer(xbuf, 288, p.ws + OFF_PRW1, p.b_pr_1, hbuf, wave, lane);
        __syncthreads();
        dense_layer(hbuf, 256, p.ws + OFF_PRW2, p.b_pr_2, h2buf, wave, lane);
        __syncthreads();
        head_z1(h2buf, p.ws + OFF_PRW3, p.b_pr_3, nullptr, p.out, nullptr,
                b0, t, wave, lane, false, 352, 384);
        __syncthreads();
    }
}

extern "C" void kernel_launch(void* const* d_in, const int* in_sizes, int n_in,
                              void* d_out, int out_size, void* d_ws, size_t ws_size,
                              hipStream_t stream) {
    (void)in_sizes; (void)n_in; (void)out_size; (void)ws_size;
    const float* emb  = (const float*)d_in[0];
    const float* act  = (const float*)d_in[1];
    const float* eps1 = (const float*)d_in[2];
    const float* eps2 = (const float*)d_in[3];
    auto Wp = [&](int i) { return (const float*)d_in[i]; };
    unsigned short* ws = (unsigned short*)d_ws;

    struct Mat { const float* W; int K, N, Kpad; unsigned off; };
    const Mat mats[15] = {
        { Wp(4),  256, 256, 256, OFF_I1W1 }, { Wp(6),  256, 256, 256, OFF_I1W2 },
        { Wp(8),  256,  64, 256, OFF_I1W3 },
        { Wp(10),  32, 256,  32, OFF_I2W1 }, { Wp(12), 256, 256, 256, OFF_I2W2 },
        { Wp(14), 256, 512, 256, OFF_I2W3 },
        { Wp(16), 518, 256, 544, OFF_POW1 }, { Wp(18), 256, 256, 256, OFF_POW2 },
        { Wp(20), 256,  64, 256, OFF_POW3 },
        { Wp(22), 294, 256, 320, OFF_Z2W1 }, { Wp(24), 256, 256, 256, OFF_Z2W2 },
        { Wp(26), 256, 512, 256, OFF_Z2W3 },
        { Wp(28), 262, 256, 288, OFF_PRW1 }, { Wp(30), 256, 256, 256, OFF_PRW2 },
        { Wp(32), 256,  64, 256, OFF_PRW3 },
    };
    for (int i = 0; i < 15; ++i) {
        int dwords = (mats[i].Kpad >> 1) * mats[i].N;
        repack_kernel<<<(dwords + 255) / 256, 256, 0, stream>>>(
            mats[i].W, ws + mats[i].off, mats[i].K, mats[i].N, mats[i].Kpad);
    }

    Params p;
    p.emb = emb; p.act = act; p.eps1 = eps1; p.eps2 = eps2;
    p.b_i1_1 = Wp(5);  p.b_i1_2 = Wp(7);  p.b_i1_3 = Wp(9);
    p.b_i2_1 = Wp(11); p.b_i2_2 = Wp(13); p.b_i2_3 = Wp(15);
    p.b_po_1 = Wp(17); p.b_po_2 = Wp(19); p.b_po_3 = Wp(21);
    p.b_z2_1 = Wp(23); p.b_z2_2 = Wp(25); p.b_z2_3 = Wp(27);
    p.b_pr_1 = Wp(29); p.b_pr_2 = Wp(31); p.b_pr_3 = Wp(33);
    p.ws = ws;
    p.out = (float*)d_out;

    slac_main<<<BATCH / 16, 256, 0, stream>>>(p);
}
